// VQAutoencoder_68771016343766
// MI455X (gfx1250) — matrix operand order repaired and checked by reading, not compiled, not run on hardware
//
#include <hip/hip_runtime.h>
#include <hip/hip_bf16.h>
#include <cstdint>
#include <cstddef>

// Problem sizes (fixed by the reference)
#define BQ     16384
#define OBSN   784
#define HDIM   512
#define DDIM   128
#define KCODES 1024
#define KP1    800   // K padding for encoder gemm1 (784 -> 800, multiple of 32)
#define NP4    832   // N padding for decoder gemm4 (784 -> 832, multiple of 64)

typedef __attribute__((ext_vector_type(16))) __bf16 v16bf;
typedef __attribute__((ext_vector_type(8)))  float  v8f;

union U16bf { uint4 u[2]; v16bf v; };

__device__ __forceinline__ __bf16 f2bf(float f) {
  uint32_t u = __builtin_bit_cast(uint32_t, f);
  uint32_t r = u + 0x7fffu + ((u >> 16) & 1u);   // round-to-nearest-even
  uint16_t s = (uint16_t)(r >> 16);
  return __builtin_bit_cast(__bf16, s);
}

// 16x32 bf16 A-fragment (row-major source, ld elements).
// ISA layout: lanes 0-15 hold row m=lane, K = {0..7, 16..23}; lanes 16-31: K = {8..15, 24..31}.
__device__ __forceinline__ v16bf load_fragA(const __bf16* tile, int ld, int kbase, int lane) {
  const int m = lane & 15, h = lane >> 4;
  const __bf16* p = tile + (size_t)m * ld + kbase + h * 8;
  U16bf r;
  r.u[0] = *(const uint4*)(p);        // K = kbase + h*8 + 0..7   -> VGPR 0..3
  r.u[1] = *(const uint4*)(p + 16);   // K = kbase + h*8 + 16..23 -> VGPR 4..7
  return r.v;
}

// 32x16 bf16 B-fragment, source is B^T row-major (n-major, ld elements).
// ISA layout: lane holds column n=lane&15, same K order as A:
// lanes 0-15: K = {0..7, 16..23}; lanes 16-31: K = {8..15, 24..31}.
__device__ __forceinline__ v16bf load_fragB(const __bf16* tile, int ld, int kbase, int lane) {
  const int n = lane & 15, h = lane >> 4;
  const __bf16* p = tile + (size_t)n * ld + kbase + h * 8;
  U16bf r;
  r.u[0] = *(const uint4*)(p);        // K = kbase + h*8 + 0..7
  r.u[1] = *(const uint4*)(p + 16);   // K = kbase + h*8 + 16..23
  return r.v;
}

#define WMMA_BF16(a, b, c) \
  __builtin_amdgcn_wmma_f32_16x16x32_bf16(false, (a), false, (b), (short)0, (c), false, false)

// ---------------------------------------------------------------------------
// Generic WMMA GEMM: C[M,N] = act(A[M,K] @ B[K,N] + bias), B given transposed.
// Wave tile 32x64 (2x4 sub-tiles), waves fully independent (L2-resident reuse).
// MODE 0: bf16 store.  MODE 1: f32 + bf16 dual store.  MODE 2: sigmoid f32
//         store with column guard n < Nreal.
// ---------------------------------------------------------------------------
template <int MODE, bool RELU>
__global__ __launch_bounds__(256) void gemm_wmma(
    const __bf16* __restrict__ A, int lda,
    const __bf16* __restrict__ Bt, int ldb,
    const float* __restrict__ bias,
    int N, int K,
    __bf16* __restrict__ Cb, float* __restrict__ Cf,
    int ldcb, int ldcf, int Nreal)
{
  const int lane = threadIdx.x & 31;
  const int wave = (blockIdx.x * 256 + threadIdx.x) >> 5;
  const int tilesN = N >> 6;
  const int row0 = (wave / tilesN) * 32;
  const int col0 = (wave % tilesN) * 64;

  v8f acc[2][4] = {};
  const __bf16* Arow0 = A + (size_t)row0 * lda;
  const __bf16* Arow1 = A + (size_t)(row0 + 16) * lda;

  for (int kc = 0; kc < K; kc += 32) {
    v16bf a0 = load_fragA(Arow0, lda, kc, lane);
    v16bf a1 = load_fragA(Arow1, lda, kc, lane);
    v16bf b0 = load_fragB(Bt + (size_t)(col0 +  0) * ldb, ldb, kc, lane);
    v16bf b1 = load_fragB(Bt + (size_t)(col0 + 16) * ldb, ldb, kc, lane);
    v16bf b2 = load_fragB(Bt + (size_t)(col0 + 32) * ldb, ldb, kc, lane);
    v16bf b3 = load_fragB(Bt + (size_t)(col0 + 48) * ldb, ldb, kc, lane);
    acc[0][0] = WMMA_BF16(a0, b0, acc[0][0]);
    acc[0][1] = WMMA_BF16(a0, b1, acc[0][1]);
    acc[0][2] = WMMA_BF16(a0, b2, acc[0][2]);
    acc[0][3] = WMMA_BF16(a0, b3, acc[0][3]);
    acc[1][0] = WMMA_BF16(a1, b0, acc[1][0]);
    acc[1][1] = WMMA_BF16(a1, b1, acc[1][1]);
    acc[1][2] = WMMA_BF16(a1, b2, acc[1][2]);
    acc[1][3] = WMMA_BF16(a1, b3, acc[1][3]);
  }

  // C/D layout: lane -> n = lane&15, rows m = (lane>>4)*8 + r (r = VGPR index)
  const int nlo = lane & 15;
  const int mhi = (lane >> 4) * 8;
#pragma unroll
  for (int mi = 0; mi < 2; mi++) {
#pragma unroll
    for (int ni = 0; ni < 4; ni++) {
      const int n_g = col0 + ni * 16 + nlo;
      float bv = 0.f;
      if (MODE != 2 || n_g < Nreal) bv = bias[n_g];
#pragma unroll
      for (int r = 0; r < 8; r++) {
        const int m_g = row0 + mi * 16 + mhi + r;
        float v = acc[mi][ni][r] + bv;
        if (RELU) v = v > 0.f ? v : 0.f;
        if constexpr (MODE == 0) {
          Cb[(size_t)m_g * ldcb + n_g] = f2bf(v);
        } else if constexpr (MODE == 1) {
          Cf[(size_t)m_g * ldcf + n_g] = v;
          Cb[(size_t)m_g * ldcb + n_g] = f2bf(v);
        } else {
          if (n_g < Nreal) {
            Cf[(size_t)m_g * ldcf + n_g] = 1.f / (1.f + __expf(-v));
          }
        }
      }
    }
  }
}

// ---------------------------------------------------------------------------
// VQ argmin: per wave, 16 rows of z (bf16) vs all K=1024 codes via WMMA.
// dist_j ~ ||E_j||^2 - 2 * z.E_j   (||z||^2 constant per row -> irrelevant)
// ---------------------------------------------------------------------------
__global__ __launch_bounds__(256) void vq_argmin_kernel(
    const __bf16* __restrict__ Zb, const __bf16* __restrict__ Eb,
    const float* __restrict__ e2, int* __restrict__ idx)
{
  const int lane = threadIdx.x & 31;
  const int wave = (blockIdx.x * 256 + threadIdx.x) >> 5;
  const int row0 = wave * 16;

  const __bf16* zt = Zb + (size_t)row0 * DDIM;
  v16bf a0 = load_fragA(zt, DDIM,  0, lane);
  v16bf a1 = load_fragA(zt, DDIM, 32, lane);
  v16bf a2 = load_fragA(zt, DDIM, 64, lane);
  v16bf a3 = load_fragA(zt, DDIM, 96, lane);

  float best[8];
  int   bidx[8];
#pragma unroll
  for (int r = 0; r < 8; r++) { best[r] = 3.4e38f; bidx[r] = 0; }

  for (int n0 = 0; n0 < KCODES; n0 += 16) {
    const __bf16* et = Eb + (size_t)n0 * DDIM;   // E is code-major: already "B^T"
    v16bf b0 = load_fragB(et, DDIM,  0, lane);
    v16bf b1 = load_fragB(et, DDIM, 32, lane);
    v16bf b2 = load_fragB(et, DDIM, 64, lane);
    v16bf b3 = load_fragB(et, DDIM, 96, lane);
    v8f acc = {};
    acc = WMMA_BF16(a0, b0, acc);
    acc = WMMA_BF16(a1, b1, acc);
    acc = WMMA_BF16(a2, b2, acc);
    acc = WMMA_BF16(a3, b3, acc);

    const int n = n0 + (lane & 15);
    const float en = e2[n];
#pragma unroll
    for (int r = 0; r < 8; r++) {
      const float d = en - 2.f * acc[r];
      if (d < best[r]) { best[r] = d; bidx[r] = n; }   // keeps lowest n per lane
    }
  }

  // Reduce across the 16 lanes sharing the same rows; lexicographic tie-break
  // to the lowest code index (matches jnp.argmin first-occurrence semantics).
#pragma unroll
  for (int r = 0; r < 8; r++) {
    float v = best[r];
    int   bi = bidx[r];
#pragma unroll
    for (int off = 8; off >= 1; off >>= 1) {
      const float ov = __shfl_xor(v, off);
      const int   oi = __shfl_xor(bi, off);
      if (ov < v || (ov == v && oi < bi)) { v = ov; bi = oi; }
    }
    bidx[r] = bi;
  }
  if ((lane & 15) == 0) {
    const int base = row0 + (lane >> 4) * 8;
#pragma unroll
    for (int r = 0; r < 8; r++) idx[base + r] = bidx[r];
  }
}

// ---------------------------------------------------------------------------
// Conversion / epilogue helpers
// ---------------------------------------------------------------------------
__global__ void conv_pad_bf16(const float* __restrict__ src, __bf16* __restrict__ dst,
                              int rows, int cols, int colsPad)
{
  const long i = (long)blockIdx.x * blockDim.x + threadIdx.x;
  if (i >= (long)rows * colsPad) return;
  const int r = (int)(i / colsPad), c = (int)(i % colsPad);
  const float v = (c < cols) ? src[(size_t)r * cols + c] : 0.f;
  dst[i] = f2bf(v);
}

// W: K x N row-major  ->  Wt: Npad x Kpad row-major (transposed, zero-padded)
__global__ void conv_wt_bf16(const float* __restrict__ W, __bf16* __restrict__ Wt,
                             int K, int N, int Kpad, int Npad)
{
  const long i = (long)blockIdx.x * blockDim.x + threadIdx.x;
  if (i >= (long)Npad * Kpad) return;
  const int n = (int)(i / Kpad), k = (int)(i % Kpad);
  const float v = (n < N && k < K) ? W[(size_t)k * N + n] : 0.f;
  Wt[i] = f2bf(v);
}

__global__ void e_norms_kernel(const float* __restrict__ E, float* __restrict__ e2)
{
  const int c = blockIdx.x * blockDim.x + threadIdx.x;
  if (c >= KCODES) return;
  float s = 0.f;
  for (int d = 0; d < DDIM; d++) { const float v = E[(size_t)c * DDIM + d]; s += v * v; }
  e2[c] = s;
}

// One block (128 threads) per row: gather z_q (f32 + bf16), idx-as-float,
// and per-row squared-error partial sums (deterministic two-stage loss).
__global__ __launch_bounds__(128) void gather_loss_kernel(
    const float* __restrict__ Zf, const float* __restrict__ E,
    const __bf16* __restrict__ Eb, const int* __restrict__ idx,
    float* __restrict__ zq_out, __bf16* __restrict__ Zqb,
    float* __restrict__ fidx, float* __restrict__ partial)
{
  const int i = blockIdx.x;
  const int d = threadIdx.x;
  const int c = idx[i];
  const float e = E[(size_t)c * DDIM + d];
  const float z = Zf[(size_t)i * DDIM + d];
  zq_out[(size_t)i * DDIM + d] = e;
  Zqb[(size_t)i * DDIM + d] = Eb[(size_t)c * DDIM + d];
  const float diff = e - z;
  float v = diff * diff;

  __shared__ float red[4];
  const int l = d & 31, w = d >> 5;
#pragma unroll
  for (int o = 16; o >= 1; o >>= 1) v += __shfl_down(v, o);
  if (l == 0) red[w] = v;
  __syncthreads();
  if (d == 0) {
    partial[i] = red[0] + red[1] + red[2] + red[3];
    fidx[i] = (float)c;
  }
}

__global__ __launch_bounds__(256) void reduce_loss_kernel(
    const float* __restrict__ partial, float* __restrict__ loss)
{
  __shared__ float sm[256];
  const int tid = threadIdx.x;
  float s = 0.f;
  for (int j = tid; j < BQ; j += 256) s += partial[j];
  sm[tid] = s;
  __syncthreads();
  for (int o = 128; o >= 1; o >>= 1) {
    if (tid < o) sm[tid] += sm[tid + o];
    __syncthreads();
  }
  // loss = codebook + 0.25*commitment = 1.25 * mean((z_q - z)^2) in forward
  if (tid == 0) loss[0] = 1.25f * sm[0] / (float)((size_t)BQ * DDIM);
}

// ---------------------------------------------------------------------------
extern "C" void kernel_launch(void* const* d_in, const int* in_sizes, int n_in,
                              void* d_out, int out_size, void* d_ws, size_t ws_size,
                              hipStream_t stream)
{
  (void)in_sizes; (void)n_in; (void)out_size; (void)ws_size;
  const float* x   = (const float*)d_in[0];
  const float* W1e = (const float*)d_in[1];
  const float* b1e = (const float*)d_in[2];
  const float* W2e = (const float*)d_in[3];
  const float* b2e = (const float*)d_in[4];
  const float* E   = (const float*)d_in[5];
  const float* W1d = (const float*)d_in[6];
  const float* b1d = (const float*)d_in[7];
  const float* W2d = (const float*)d_in[8];
  const float* b2d = (const float*)d_in[9];

  // Output layout: x_hat [B*OBS] | z_q_st [B*D] | loss [1] | idx [B]
  float* out_xhat = (float*)d_out;
  float* out_zq   = out_xhat + (size_t)BQ * OBSN;
  float* out_loss = out_zq + (size_t)BQ * DDIM;
  float* out_idx  = out_loss + 1;

  // Workspace carve-up (256-B aligned)
  char* ws = (char*)d_ws;
  size_t off = 0;
  auto carve = [&](size_t bytes) -> void* {
    void* p = ws + off;
    off = (off + bytes + 255) & ~(size_t)255;
    return p;
  };
  __bf16* Xb      = (__bf16*)carve((size_t)BQ * KP1 * 2);     // 26.2 MB (reused as Hd)
  __bf16* W1et    = (__bf16*)carve((size_t)HDIM * KP1 * 2);
  __bf16* H1      = (__bf16*)carve((size_t)BQ * HDIM * 2);    // 16.8 MB
  __bf16* W2et    = (__bf16*)carve((size_t)DDIM * HDIM * 2);
  float*  Zf      = (float*) carve((size_t)BQ * DDIM * 4);    // 8.4 MB
  __bf16* Zb      = (__bf16*)carve((size_t)BQ * DDIM * 2);    // reused as Zqb
  __bf16* Eb      = (__bf16*)carve((size_t)KCODES * DDIM * 2);
  float*  e2      = (float*) carve((size_t)KCODES * 4);
  int*    idxi    = (int*)   carve((size_t)BQ * 4);
  __bf16* W1dt    = (__bf16*)carve((size_t)HDIM * DDIM * 2);
  __bf16* W2dt    = (__bf16*)carve((size_t)NP4 * HDIM * 2);
  float*  partial = (float*) carve((size_t)BQ * 4);
  __bf16* Hd  = Xb;   // alias: x no longer needed after gemm1
  __bf16* Zqb = Zb;   // alias: Zb no longer needed after vq_argmin

  auto cdiv = [](long a, long b) { return (unsigned)((a + b - 1) / b); };

  // --- bf16 conversions (weights transposed; K/N padding zero-filled) ---
  conv_pad_bf16<<<cdiv((long)BQ * KP1, 256), 256, 0, stream>>>(x, Xb, BQ, OBSN, KP1);
  conv_wt_bf16 <<<cdiv((long)HDIM * KP1, 256), 256, 0, stream>>>(W1e, W1et, OBSN, HDIM, KP1, HDIM);
  conv_wt_bf16 <<<cdiv((long)DDIM * HDIM, 256), 256, 0, stream>>>(W2e, W2et, HDIM, DDIM, HDIM, DDIM);
  conv_pad_bf16<<<cdiv((long)KCODES * DDIM, 256), 256, 0, stream>>>(E, Eb, KCODES, DDIM, DDIM);
  e_norms_kernel<<<4, 256, 0, stream>>>(E, e2);
  conv_wt_bf16 <<<cdiv((long)HDIM * DDIM, 256), 256, 0, stream>>>(W1d, W1dt, DDIM, HDIM, DDIM, HDIM);
  conv_wt_bf16 <<<cdiv((long)NP4 * HDIM, 256), 256, 0, stream>>>(W2d, W2dt, HDIM, OBSN, HDIM, NP4);

  // --- Encoder ---
  // h = relu(x @ W1e + b1e): M=16384 N=512 K=800 -> waves=(512)*(8)=4096 -> 512 blocks
  gemm_wmma<0, true><<<512, 256, 0, stream>>>(Xb, KP1, W1et, KP1, b1e,
                                              HDIM, KP1, H1, nullptr, HDIM, 0, HDIM);
  // z = h @ W2e + b2e (f32 + bf16): N=128 K=512 -> waves=512*2=1024 -> 128 blocks
  gemm_wmma<1, false><<<128, 256, 0, stream>>>(H1, HDIM, W2et, HDIM, b2e,
                                               DDIM, HDIM, Zb, Zf, DDIM, DDIM, DDIM);

  // --- VQ: argmin over codebook ---  waves = 16384/16 = 1024 -> 128 blocks
  vq_argmin_kernel<<<128, 256, 0, stream>>>(Zb, Eb, e2, idxi);

  // --- Gather z_q, idx, loss partials; deterministic reduction ---
  gather_loss_kernel<<<BQ, 128, 0, stream>>>(Zf, E, Eb, idxi, out_zq, Zqb, out_idx, partial);
  reduce_loss_kernel<<<1, 256, 0, stream>>>(partial, out_loss);

  // --- Decoder ---
  // hd = relu(z_q @ W1d + b1d): N=512 K=128 -> 4096 waves -> 512 blocks
  gemm_wmma<0, true><<<512, 256, 0, stream>>>(Zqb, DDIM, W1dt, DDIM, b1d,
                                              HDIM, DDIM, Hd, nullptr, HDIM, 0, HDIM);
  // x_hat = sigmoid(hd @ W2d + b2d): N=832(pad) K=512, store guard n<784
  // waves = 512*13 = 6656 -> 832 blocks
  gemm_wmma<2, false><<<832, 256, 0, stream>>>(Hd, HDIM, W2dt, HDIM, b2d,
                                               NP4, HDIM, nullptr, out_xhat, 0, OBSN, OBSN);
}
